// WindowAttention_38826504356178
// MI455X (gfx1250) — matrix operand order repaired and checked by reading, not compiled, not run on hardware
//
#include <hip/hip_runtime.h>
#include <hip/hip_bf16.h>

#define NTOK 64
#define DIM  256
#define NH   8
#define HD   32
#define LDX  264   // padded row stride (bf16 elems) for [64][256] tiles
#define LDP  72    // padded stride for P [64][64] per head
#define LDV  72    // padded stride for vT [256][64]

typedef __attribute__((ext_vector_type(16))) __bf16 v16bf;
typedef __attribute__((ext_vector_type(2)))  __bf16 v2bf;
typedef __attribute__((ext_vector_type(8)))  float  v8f;

union FragU { v16bf v; uint4 q[2]; };

// hardware f32 -> bf16 (RNE) converts
static __device__ __forceinline__ unsigned pack2bf(float lo, float hi) {
#if __has_builtin(__builtin_amdgcn_cvt_pk_bf16_f32)
  union { v2bf v; unsigned u; } c;
  c.v = __builtin_amdgcn_cvt_pk_bf16_f32(lo, hi);
  return c.u;
#else
  union { __bf16 h[2]; unsigned u; } c;
  c.h[0] = (__bf16)lo; c.h[1] = (__bf16)hi;
  return c.u;
#endif
}

static __device__ __forceinline__ unsigned short f2bf(float f) {
  union { __bf16 h; unsigned short u; } c;
  c.h = (__bf16)f;
  return c.u;
}

// A fragment (16x32 bf16): lane l16 = M row; elems 0..7 = K[k0+half*8 .. +7],
// elems 8..15 = K[k0+half*8+16 .. +23]. Source stored row-major [M][ld].
static __device__ __forceinline__ v16bf frag_A(const unsigned short* base, int ld,
                                               int row, int k0, int half) {
  FragU f;
  const unsigned short* p = base + row * ld + k0 + half * 8;
  f.q[0] = *(const uint4*)(p);
  f.q[1] = *(const uint4*)(p + 16);
  return f.v;
}

// B fragment (32x16 bf16): lane l16 = N col; elems 0..7 = K[k0+half*8 .. +7], elems 8..15 = K[k0+half*8+16 .. +23].
// Source stored "per-column K-contiguous": base[col*ld + k].
static __device__ __forceinline__ v16bf frag_B(const unsigned short* base, int ld,
                                               int col, int k0, int half) {
  FragU f;
  const unsigned short* p = base + col * ld + k0 + half * 8;
  f.q[0] = *(const uint4*)(p);
  f.q[1] = *(const uint4*)(p + 16);
  return f.v;
}

static __device__ __forceinline__ v8f wmma_bf16(v16bf a, v16bf b, v8f c) {
  return __builtin_amdgcn_wmma_f32_16x16x32_bf16(false, a, false, b,
                                                 (short)0, c, false, false);
}

// ---------------- prep kernels (run once per launch, tiny) ----------------

__global__ void prep_weights_bf16(const float* __restrict__ q_w,
                                  const float* __restrict__ kv_w,
                                  const float* __restrict__ proj_w,
                                  unsigned short* __restrict__ qwB,
                                  unsigned short* __restrict__ kvwB,
                                  unsigned short* __restrict__ projB) {
  int i = blockIdx.x * blockDim.x + threadIdx.x;       // 0 .. 131071
  if (i < 65536)  qwB[i]   = f2bf(q_w[i]);
  if (i < 131072) kvwB[i]  = f2bf(kv_w[i]);
  if (i < 65536)  projB[i] = f2bf(proj_w[i]);
}

__global__ void prep_bias(const float* __restrict__ rpb, float* __restrict__ biasP) {
  int t = blockIdx.x * blockDim.x + threadIdx.x;       // 0 .. 32767  [h][i][j]
  int h = t >> 12, ij = t & 4095, i = ij >> 6, j = ij & 63;
  int ri = i >> 3, ci = i & 7, rj = j >> 3, cj = j & 7;
  int idx = (ri - rj + 7) * 15 + (ci - cj + 7);        // (2*WS-1)=15
  biasP[t] = rpb[idx * NH + h];
}

// ---------------- main fused window-attention kernel ----------------

__global__ void __launch_bounds__(256)
win_attn_kernel(const float* __restrict__ ref, const float* __restrict__ adj,
                const float* __restrict__ mask,
                const float* __restrict__ q_b, const float* __restrict__ kv_b,
                const float* __restrict__ proj_b,
                const unsigned short* __restrict__ qwB,
                const unsigned short* __restrict__ kvwB,
                const unsigned short* __restrict__ projB,
                const float* __restrict__ biasP,
                float* __restrict__ out) {
  extern __shared__ unsigned short smem[];
  // buf0: 36864 ush: refB[64][LDX] + adjB[64][LDX]  -> later pB[8][64][LDP]
  // buf1: 16896 ush: qB[64][LDX]                    -> later outB[64][LDX]
  // buf2: 16896 ush: kB[64][LDX]
  // buf3: 18432 ush: vT[256][LDV]
  unsigned short* refB = smem;
  unsigned short* adjB = smem + 16896;
  unsigned short* pB   = smem;                 // overlays ref/adj after barrier
  unsigned short* qB   = smem + 36864;
  unsigned short* outB = qB;                   // same columns per wave: safe
  unsigned short* kB   = smem + 36864 + 16896;
  unsigned short* vT   = smem + 36864 + 2 * 16896;

  const int b    = blockIdx.x;
  const int tid  = threadIdx.x;
  const int wave = tid >> 5;                   // = head index
  const int lane = tid & 31;
  const int half = lane >> 4;
  const int l16  = lane & 15;
  const float scale = 0.17677669529663687f;    // 1/sqrt(32)

  // weight prefetch hint (L2-resident across all 4096 blocks)
  if (tid < 32) {
    __builtin_prefetch(qwB + tid * 4096, 0, 1);
    __builtin_prefetch(kvwB + tid * 8192, 0, 1);
    __builtin_prefetch(projB + tid * 4096, 0, 1);
  }

  // ---- Phase 1: load ref/adj window, convert f32 -> bf16 into LDS ----
  {
    const float* rp = ref + (size_t)b * (NTOK * DIM);
    const float* ap = adj + (size_t)b * (NTOK * DIM);
#pragma unroll
    for (int it = 0; it < 8; ++it) {
      int e   = (it * 256 + tid) * 8;          // flat f32 index, 8-elem granule
      int row = e >> 8, col = e & 255;
      float4 r0 = *(const float4*)(rp + e);
      float4 r1 = *(const float4*)(rp + e + 4);
      float4 a0 = *(const float4*)(ap + e);
      float4 a1 = *(const float4*)(ap + e + 4);
      uint4 rk, ak;
      rk.x = pack2bf(r0.x, r0.y); rk.y = pack2bf(r0.z, r0.w);
      rk.z = pack2bf(r1.x, r1.y); rk.w = pack2bf(r1.z, r1.w);
      ak.x = pack2bf(a0.x, a0.y); ak.y = pack2bf(a0.z, a0.w);
      ak.z = pack2bf(a1.x, a1.y); ak.w = pack2bf(a1.z, a1.w);
      *(uint4*)(refB + row * LDX + col) = rk;
      *(uint4*)(adjB + row * LDX + col) = ak;
    }
  }
  __syncthreads();

  // ---- Phase 2: q = ref @ q_w^T + q_b  (wave w -> columns of head w) ----
  {
    const int h = wave;
#pragma unroll
    for (int nt = 0; nt < 2; ++nt) {
      int n0 = h * HD + nt * 16;
      float bias = q_b[n0 + l16];
      v8f acc[4];
#pragma unroll
      for (int mt = 0; mt < 4; ++mt)
#pragma unroll
        for (int r = 0; r < 8; ++r) acc[mt][r] = bias;
#pragma unroll
      for (int kt = 0; kt < 8; ++kt) {
        v16bf bf = frag_B(qwB, DIM, n0 + l16, kt * 32, half);
#pragma unroll
        for (int mt = 0; mt < 4; ++mt) {
          v16bf af = frag_A(refB, LDX, mt * 16 + l16, kt * 32, half);
          acc[mt] = wmma_bf16(af, bf, acc[mt]);
        }
      }
#pragma unroll
      for (int mt = 0; mt < 4; ++mt)
#pragma unroll
        for (int r = 0; r < 8; ++r)
          qB[(mt * 16 + half * 8 + r) * LDX + n0 + l16] = f2bf(acc[mt][r] * scale);
    }
  }

  // ---- Phase 3: kv = adj @ kv_w^T + kv_b (wave w -> out cols [64w,64w+64)) ----
  {
#pragma unroll
    for (int nt = 0; nt < 4; ++nt) {
      int co = wave * 64 + nt * 16 + l16;      // 0..511
      float bias = kv_b[co];
      v8f acc[4];
#pragma unroll
      for (int mt = 0; mt < 4; ++mt)
#pragma unroll
        for (int r = 0; r < 8; ++r) acc[mt][r] = bias;
#pragma unroll
      for (int kt = 0; kt < 8; ++kt) {
        v16bf bf = frag_B(kvwB, DIM, co, kt * 32, half);
#pragma unroll
        for (int mt = 0; mt < 4; ++mt) {
          v16bf af = frag_A(adjB, LDX, mt * 16 + l16, kt * 32, half);
          acc[mt] = wmma_bf16(af, bf, acc[mt]);
        }
      }
      if (co < 256) {                           // K: store [token][dim]
#pragma unroll
        for (int mt = 0; mt < 4; ++mt)
#pragma unroll
          for (int r = 0; r < 8; ++r)
            kB[(mt * 16 + half * 8 + r) * LDX + co] = f2bf(acc[mt][r]);
      } else {                                  // V: store transposed [dim][token]
        int vd = co - 256;
#pragma unroll
        for (int mt = 0; mt < 4; ++mt) {
          uint4 pk;
          pk.x = pack2bf(acc[mt][0], acc[mt][1]);
          pk.y = pack2bf(acc[mt][2], acc[mt][3]);
          pk.z = pack2bf(acc[mt][4], acc[mt][5]);
          pk.w = pack2bf(acc[mt][6], acc[mt][7]);
          *(uint4*)(vT + vd * LDV + mt * 16 + half * 8) = pk;
        }
      }
    }
  }
  __syncthreads();   // kB/vT visible to all waves; ref/adj space freed for pB

  // ---- Phase 4: S = q_h k_h^T + bias + mask; row softmax; P -> LDS bf16 ----
  {
    const int h = wave;
    const float* maskP = mask + (size_t)(b & 1023) * 4096;   // nW = 1024
    const float* biasH = biasP + h * 4096;
    v8f s[4][4];
    v16bf bk[4];
#pragma unroll
    for (int nt = 0; nt < 4; ++nt)
      bk[nt] = frag_B(kB, LDX, nt * 16 + l16, h * HD, half);
#pragma unroll
    for (int mt = 0; mt < 4; ++mt) {
      v16bf aq = frag_A(qB, LDX, mt * 16 + l16, h * HD, half);
#pragma unroll
      for (int nt = 0; nt < 4; ++nt) {
        v8f c;
#pragma unroll
        for (int r = 0; r < 8; ++r) {
          int i = mt * 16 + half * 8 + r, j = nt * 16 + l16;
          c[r] = biasH[i * 64 + j] + maskP[i * 64 + j];
        }
        s[mt][nt] = wmma_bf16(aq, bk[nt], c);
      }
    }
    unsigned short* ph = pB + h * (64 * LDP);
#pragma unroll
    for (int mt = 0; mt < 4; ++mt) {
#pragma unroll
      for (int r = 0; r < 8; ++r) {
        float m = fmaxf(fmaxf(s[mt][0][r], s[mt][1][r]),
                        fmaxf(s[mt][2][r], s[mt][3][r]));
#pragma unroll
        for (int d = 1; d < 16; d <<= 1) m = fmaxf(m, __shfl_xor(m, d, 16));
        float sum = 0.f;
#pragma unroll
        for (int nt = 0; nt < 4; ++nt) {
          float p = __expf(s[mt][nt][r] - m);
          s[mt][nt][r] = p;
          sum += p;
        }
#pragma unroll
        for (int d = 1; d < 16; d <<= 1) sum += __shfl_xor(sum, d, 16);
        float inv = __builtin_amdgcn_rcpf(sum);   // softmax normalizer, fast rcp
#pragma unroll
        for (int nt = 0; nt < 4; ++nt)
          ph[(mt * 16 + half * 8 + r) * LDP + nt * 16 + l16] =
              f2bf(s[mt][nt][r] * inv);
      }
    }
  }

  // ---- Phase 5: out_h = P @ v_h -> outB[:, 32h:32h+32] (same-wave region) ----
  {
    const int h = wave;
    const unsigned short* ph = pB + h * (64 * LDP);
#pragma unroll
    for (int nt = 0; nt < 2; ++nt) {
      v8f acc[4];
#pragma unroll
      for (int mt = 0; mt < 4; ++mt)
#pragma unroll
        for (int r = 0; r < 8; ++r) acc[mt][r] = 0.f;
#pragma unroll
      for (int kt = 0; kt < 2; ++kt) {
        v16bf bv = frag_B(vT, LDV, h * HD + nt * 16 + l16, kt * 32, half);
#pragma unroll
        for (int mt = 0; mt < 4; ++mt) {
          v16bf ap = frag_A(ph, LDP, mt * 16 + l16, kt * 32, half);
          acc[mt] = wmma_bf16(ap, bv, acc[mt]);
        }
      }
#pragma unroll
      for (int mt = 0; mt < 4; ++mt)
#pragma unroll
        for (int r = 0; r < 8; ++r)
          outB[(mt * 16 + half * 8 + r) * LDX + h * HD + nt * 16 + l16] =
              f2bf(acc[mt][r]);
    }
  }
  __syncthreads();   // all heads' outputs visible for projection

  // ---- Phase 6: final = outB @ proj_w^T + proj_b ----
  {
    float* op = out + (size_t)b * (NTOK * DIM);
#pragma unroll
    for (int nt = 0; nt < 2; ++nt) {
      int n0 = wave * HD + nt * 16;
      float bias = proj_b[n0 + l16];
      v8f acc[4];
#pragma unroll
      for (int mt = 0; mt < 4; ++mt)
#pragma unroll
        for (int r = 0; r < 8; ++r) acc[mt][r] = bias;
#pragma unroll
      for (int kt = 0; kt < 8; ++kt) {
        v16bf bf = frag_B(projB, DIM, n0 + l16, kt * 32, half);
#pragma unroll
        for (int mt = 0; mt < 4; ++mt) {
          v16bf af = frag_A(outB, LDX, mt * 16 + l16, kt * 32, half);
          acc[mt] = wmma_bf16(af, bf, acc[mt]);
        }
      }
#pragma unroll
      for (int mt = 0; mt < 4; ++mt)
#pragma unroll
        for (int r = 0; r < 8; ++r)
          op[(mt * 16 + half * 8 + r) * DIM + n0 + l16] = acc[mt][r];
    }
  }
}

extern "C" void kernel_launch(void* const* d_in, const int* in_sizes, int n_in,
                              void* d_out, int out_size, void* d_ws, size_t ws_size,
                              hipStream_t stream) {
  const float* ref    = (const float*)d_in[0];
  const float* adj    = (const float*)d_in[1];
  const float* mask   = (const float*)d_in[2];
  const float* q_w    = (const float*)d_in[3];
  const float* q_b    = (const float*)d_in[4];
  const float* kv_w   = (const float*)d_in[5];
  const float* kv_b   = (const float*)d_in[6];
  const float* proj_w = (const float*)d_in[7];
  const float* proj_b = (const float*)d_in[8];
  const float* rpb    = (const float*)d_in[9];
  float* out = (float*)d_out;

  char* ws = (char*)d_ws;
  unsigned short* qwB   = (unsigned short*)(ws);            // 128 KB
  unsigned short* kvwB  = (unsigned short*)(ws + 131072);   // 256 KB
  unsigned short* projB = (unsigned short*)(ws + 393216);   // 128 KB
  float*          biasP = (float*)(ws + 524288);            // 128 KB

  prep_weights_bf16<<<512, 256, 0, stream>>>(q_w, kv_w, proj_w, qwB, kvwB, projB);
  prep_bias<<<128, 256, 0, stream>>>(rpb, biasP);

  size_t shmem = (size_t)(36864 + 16896 + 16896 + 18432) * sizeof(unsigned short);
  hipFuncSetAttribute((const void*)win_attn_kernel,
                      hipFuncAttributeMaxDynamicSharedMemorySize, (int)shmem);
  win_attn_kernel<<<4096, 256, shmem, stream>>>(ref, adj, mask, q_b, kv_b, proj_b,
                                                qwB, kvwB, projB, biasP, out);
}
